// ProgramDecoder_6236292513844
// MI455X (gfx1250) — compile-verified, matrix operand order checked by reading, not run on hardware
//
#include <hip/hip_runtime.h>
#include <hip/hip_bf16.h>
#include <math.h>

// Problem constants (from the reference)
#define BB 8192
#define TT 30
#define VV 128
#define EE 512
#define DD 512
#define KDIM 1024          // D + E
#define KD 512             // K-depth of both WMMA GEMMs
#define KTILES 16          // KD / 32
#define LDSS 40            // padded LDS row stride (bf16): 80B rows, 16B-divisible, conflict-free
#define SCALE 0.044194173824159216f  // 1/sqrt(512)

typedef __attribute__((ext_vector_type(16))) __bf16 v16bf;
typedef __attribute__((ext_vector_type(8)))  __bf16 v8bf;
typedef __attribute__((ext_vector_type(8)))  float  v8f;
typedef __attribute__((ext_vector_type(4)))  unsigned int u32x4;
typedef __attribute__((ext_vector_type(8)))  unsigned int u32x8;

// ---------------------------------------------------------------------------
// CDNA5 data-movement helpers
// ---------------------------------------------------------------------------

// Generic __shared__ pointer -> workgroup-relative LDS byte offset
// (LDS aperture: the LDS offset is the low 32 address bits)
__device__ __forceinline__ unsigned lds_off(const void* p) {
    return (unsigned)(unsigned long long)p;
}

// TDM: 2D tile load Global->LDS.  D# group0/group1 built in SGPRs.
//  g1w0 packs: data_size(2B)=1<<16 | pad_enable<<20 | pad_interval<<22 | pad_amount<<25
//  tile01 = (tile_dim0 << 16) | tensor_dim1[31:16]   (tensor dims set to 2^20: no OOB)
__device__ __forceinline__ void tdm_load_2d(unsigned lds_addr, const void* gaddr,
                                            unsigned g1w0, unsigned tile01,
                                            unsigned tile_dim1, unsigned stride_elems) {
    unsigned long long ga = (unsigned long long)gaddr;
    u32x4 g0 = { 1u,                                   // count=1, is_restore=0, gather=0
                 lds_addr,                             // lds_addr [63:32]
                 (unsigned)ga,                         // global_addr [95:64]
                 ((unsigned)(ga >> 32) & 0x01FFFFFFu) | 0x80000000u }; // addr hi | type=2
    u32x8 g1 = { g1w0,                                 // mask=0, data_size, padding cfg
                 0u,                                   // atomic_barrier=0, tensor_dim0 lo16=0
                 0x10u,                                // tensor_dim0 hi16 = 0x10 (2^20)
                 tile01,                               // tensor_dim1 hi16 | tile_dim0
                 tile_dim1,                            // tile_dim1 (tile_dim2=0)
                 stride_elems,                         // tensor_dim0_stride lo32
                 0u, 0u };
    asm volatile("tensor_load_to_lds %0, %1" :: "s"(g0), "s"(g1) : "memory");
}

// Async global->LDS, 16B per lane (one wave = one 512B mask row)
__device__ __forceinline__ void async_b128(unsigned lds_addr, unsigned voff, const void* saddr) {
    asm volatile("global_load_async_to_lds_b128 %0, %1, %2"
                 :: "v"(lds_addr), "v"(voff), "s"(saddr) : "memory");
}
__device__ __forceinline__ void wait_async_le1() {
    asm volatile("s_wait_asynccnt 0x1" ::: "memory");
}
__device__ __forceinline__ void wait_async_le0() {
    asm volatile("s_wait_asynccnt 0x0" ::: "memory");
}

// ---------------------------------------------------------------------------
// Small dense precompute kernels (tiny: <=131072 outputs, K=512 dots)
// ---------------------------------------------------------------------------

// k[v,e] = emb[v,:] . Wk[e,:] + bk[e]        (k = emb @ Wk^T + bk)
__global__ void kmat_kernel(const float* __restrict__ emb, const float* __restrict__ Wk,
                            const float* __restrict__ bk, float* __restrict__ kmat) {
    int idx = blockIdx.x * blockDim.x + threadIdx.x;    // 65536
    int v = idx >> 9, e = idx & 511;
    const float* a = emb + (size_t)v * EE;
    const float* w = Wk  + (size_t)e * EE;
    float s = bk[e];
    #pragma unroll 4
    for (int j = 0; j < EE; ++j) s += a[j] * w[j];
    kmat[idx] = s;
}

// K2[v,j] = sum_e k[v,e] * Wq[e,j]           (K2 = k @ Wq)
__global__ void k2_kernel(const float* __restrict__ kmat, const float* __restrict__ Wq,
                          float* __restrict__ K2) {
    int idx = blockIdx.x * blockDim.x + threadIdx.x;    // 65536
    int v = idx >> 9, j = idx & 511;
    const float* a = kmat + (size_t)v * EE;
    float s = 0.f;
    #pragma unroll 4
    for (int e = 0; e < EE; ++e) s += a[e] * Wq[(size_t)e * EE + j];   // coalesced on Wq
    K2[idx] = s;
}

// W3s[v,d] = scale * sum_j K2[v,j] * W_lin[j,d]   (+ bf16 copy for WMMA)
__global__ void w3_kernel(const float* __restrict__ K2, const float* __restrict__ Wlin,
                          float* __restrict__ W3s, __bf16* __restrict__ W3bf) {
    int idx = blockIdx.x * blockDim.x + threadIdx.x;    // 131072
    int v = idx >> 10, d = idx & 1023;
    const float* a = K2 + (size_t)v * EE;
    float s = 0.f;
    #pragma unroll 4
    for (int j = 0; j < EE; ++j) s += a[j] * Wlin[(size_t)j * KDIM + d]; // coalesced on Wlin
    s *= SCALE;
    W3s[idx]  = s;
    W3bf[idx] = (__bf16)s;
}

// c[v] = scale * ( b_lin . K2[v,:] + bq . k[v,:] )
__global__ void cvec_kernel(const float* __restrict__ K2, const float* __restrict__ kmat,
                            const float* __restrict__ b_lin, const float* __restrict__ bq,
                            float* __restrict__ cvec) {
    int v = threadIdx.x;                                 // 128 threads
    const float* k2r = K2   + (size_t)v * EE;
    const float* kr  = kmat + (size_t)v * EE;
    float s = 0.f;
    for (int j = 0; j < EE; ++j) s += b_lin[j] * k2r[j] + bq[j] * kr[j];
    cvec[v] = s * SCALE;
}

// ---------------------------------------------------------------------------
// WMMA bf16 GEMM:  C[M,128] = A[M,512](f32->bf16) @ W[128, ldw](bf16, cols colOff..+511)^T
// Block = 256 threads (8 wave32s); 128x128 C tile, K=512 in 16 steps of 32.
// B tile staged by the Tensor Data Mover (pads to stride-40 bf16 rows itself);
// A tile staged by VALU (needs f32->bf16 conversion) overlapping the DMA.
// Fragment layout per CDNA5 16-bit A/B VGPR tables:
//   lane<16 : K = kb+0..7, kb+16..23 (kb=0);  lane>=16: same with kb=8.
// ---------------------------------------------------------------------------
__device__ __forceinline__ v16bf frag_rowK(const __bf16* row, int kbl) {
    v8bf lo = *(const v8bf*)(row + kbl);        // 16B aligned (LDSS*2 = 80B rows)
    v8bf hi = *(const v8bf*)(row + kbl + 16);
    return __builtin_shufflevector(lo, hi, 0,1,2,3,4,5,6,7,8,9,10,11,12,13,14,15);
}

__global__ void __launch_bounds__(256)
gemm_bf16_128(const float* __restrict__ A, const __bf16* __restrict__ W,
              int ldw, int colOff, float* __restrict__ C) {
    __shared__ __align__(16) __bf16 Alds[128 * LDSS];
    __shared__ __align__(16) __bf16 Blds[128 * LDSS];

    const int tid  = threadIdx.x;
    const int wv   = tid >> 5;
    const int lane = tid & 31;
    const int rowBase = blockIdx.x * 128;

    const int mrow = wv * 16 + (lane & 15);     // A row this lane feeds
    const int kbl  = (lane >> 4) << 3;          // 0 or 8
    const unsigned bldsBase = lds_off(&Blds[0]);

    // TDM group1 word0: data_size=2B (1<<16) | pad_enable (1<<20)
    //                 | pad_interval=16DW/64B (3<<22) | pad_amount=4DW/16B (3<<25)
    const unsigned g1w0   = (1u << 16) | (1u << 20) | (3u << 22) | (3u << 25);
    const unsigned tile01 = (32u << 16) | 0x10u;   // tile_dim0=32 | tensor_dim1 hi16

    v8f acc[8];
    #pragma unroll
    for (int n = 0; n < 8; ++n) acc[n] = (v8f){0.f,0.f,0.f,0.f,0.f,0.f,0.f,0.f};

    for (int kt = 0; kt < KTILES; ++kt) {
        const int k0 = kt * 32;
        // Kick the TDM for the 128x32 bf16 B tile (wave 0 issues once).
        if (wv == 0)
            tdm_load_2d(bldsBase, W + (size_t)colOff + k0, g1w0, tile01, 128u, (unsigned)ldw);
        // Meanwhile: cooperative A stage, f32 -> bf16, 128x32.
        for (int i = tid; i < 128 * 32; i += 256) {
            int r = i >> 5, kk = i & 31;
            Alds[r * LDSS + kk] = (__bf16)A[(size_t)(rowBase + r) * KD + k0 + kk];
        }
        if (wv == 0) __builtin_amdgcn_s_wait_tensorcnt(0);
        __syncthreads();

        v16bf af = frag_rowK(&Alds[mrow * LDSS], kbl);
        #pragma unroll
        for (int n = 0; n < 8; ++n) {
            v16bf bf = frag_rowK(&Blds[(n * 16 + (lane & 15)) * LDSS], kbl);
            acc[n] = __builtin_amdgcn_wmma_f32_16x16x32_bf16(
                false, af, false, bf, (short)0, acc[n], false, false);
        }
        __syncthreads();
    }

    // C/D layout: VGPR r -> M = r (+8 for lanes>=16), N = lane&15
    const int mloc = wv * 16 + ((lane >> 4) << 3);
    const int nloc = lane & 15;
    #pragma unroll
    for (int n = 0; n < 8; ++n)
        #pragma unroll
        for (int r = 0; r < 8; ++r)
            C[(size_t)(rowBase + mloc + r) * VV + n * 16 + nloc] = acc[n][r];
}

// ---------------------------------------------------------------------------
// Streaming pass (HBM roofline: reads the 126MB mask).  One block per batch
// row; each wave owns mask rows t = wv, wv+8, ... and double-buffers them
// into wave-private LDS via global_load_async_to_lds_b128 (ASYNCcnt is
// per-wave and completion is in-order, so s_wait_asynccnt 1 releases the
// current slot while the next row is still in flight).
// ---------------------------------------------------------------------------
__global__ void __launch_bounds__(256)
finalize_kernel(const float* __restrict__ Senc, const float* __restrict__ Stok,
                const float* __restrict__ cvec, const float* __restrict__ mask,
                const int* __restrict__ parent, const int* __restrict__ targets,
                float* __restrict__ out) {
    const int b    = blockIdx.x;
    const int tid  = threadIdx.x;
    const int wv   = tid >> 5;
    const int lane = tid & 31;

    __shared__ __align__(16) float maskbuf[8][2][VV];   // 8KB: per-wave double buffer
    __shared__ float senc_s[VV];
    __shared__ float c_s[VV];
    __shared__ float partial[8];

    if (tid < VV) { senc_s[tid] = Senc[(size_t)b * VV + tid]; c_s[tid] = cvec[tid]; }
    if (tid < 8) partial[tid] = 0.f;
    __syncthreads();

    const float* base = mask + (size_t)b * TT * VV;
    const unsigned voff  = (unsigned)lane * 16u;
    const unsigned lbuf0 = lds_off(&maskbuf[wv][0][0]) + voff;
    const unsigned lbuf1 = lds_off(&maskbuf[wv][1][0]) + voff;

    // Prime slot 0 with this wave's first row.
    async_b128(lbuf0, voff, base + (size_t)wv * VV);

    float wsum = 0.f;
    int it = 0;
    for (int t = wv; t < TT; t += 8, ++it) {
        const int slot = it & 1;
        const bool more = (t + 8) < TT;
        if (more)
            async_b128(slot ? lbuf0 : lbuf1, voff, base + (size_t)(t + 8) * VV);
        if (more) wait_async_le1(); else wait_async_le0();

        const int p  = parent [b * TT + t];
        const int tg = targets[b * TT + t];
        const float* mrow = &maskbuf[wv][slot][0];      // LDS
        const float* srow = Stok + (size_t)p * VV;      // L2-resident 64KB table

        float s[4], m = -3.0e38f, stgt = 0.f;
        #pragma unroll
        for (int j = 0; j < 4; ++j) {
            int v = lane + 32 * j;
            s[j] = senc_s[v] + srow[v] + c_s[v] + mrow[v];
            m = fmaxf(m, s[j]);
            if (v == tg) stgt = s[j];
        }
        #pragma unroll
        for (int off = 16; off > 0; off >>= 1) m = fmaxf(m, __shfl_xor(m, off));
        float e = 0.f;
        #pragma unroll
        for (int j = 0; j < 4; ++j) e += __expf(s[j] - m);
        #pragma unroll
        for (int off = 16; off > 0; off >>= 1) e += __shfl_xor(e, off);
        #pragma unroll
        for (int off = 16; off > 0; off >>= 1) stgt += __shfl_xor(stgt, off);
        wsum += stgt - (m + __logf(e));                 // log_softmax at target
    }
    if (lane == 0) partial[wv] = wsum;
    __syncthreads();
    if (tid == 0) {
        float tot = 0.f;
        #pragma unroll
        for (int i = 0; i < 8; ++i) tot += partial[i];
        out[b] = tot;
    }
}

// ---------------------------------------------------------------------------
// Host-side launch
// ---------------------------------------------------------------------------
extern "C" void kernel_launch(void* const* d_in, const int* in_sizes, int n_in,
                              void* d_out, int out_size, void* d_ws, size_t ws_size,
                              hipStream_t stream) {
    (void)in_sizes; (void)n_in; (void)out_size; (void)ws_size;

    const float* enc    = (const float*)d_in[0];   // [B, D]
    const int*   parent = (const int*)  d_in[1];   // [B, T]
    const int*   targ   = (const int*)  d_in[2];   // [B, T]
    const float* mask   = (const float*)d_in[3];   // [B, T, V]
    const float* emb    = (const float*)d_in[4];   // [V, E]
    const float* Wlin   = (const float*)d_in[5];   // [E, D+E]
    const float* b_lin  = (const float*)d_in[6];   // [E]
    const float* Wq     = (const float*)d_in[7];   // [E, E]
    const float* bq     = (const float*)d_in[8];   // [E]
    const float* Wk     = (const float*)d_in[9];   // [E, E]
    const float* bk     = (const float*)d_in[10];  // [E]
    float* out = (float*)d_out;                    // [B]

    // Workspace layout (~5.6 MB)
    float*  kmat = (float*)d_ws;                   // V*E
    float*  K2   = kmat + (size_t)VV * EE;         // V*E
    float*  W3s  = K2   + (size_t)VV * EE;         // V*KDIM
    float*  cvec = W3s  + (size_t)VV * KDIM;       // V (padded to 128)
    float*  Stok = cvec + 128;                     // V*V
    float*  Senc = Stok + (size_t)VV * VV;         // B*V
    __bf16* W3bf = (__bf16*)(Senc + (size_t)BB * VV); // V*KDIM bf16

    // 1) k = emb @ Wk^T + bk
    kmat_kernel<<<256, 256, 0, stream>>>(emb, Wk, bk, kmat);
    // 2) K2 = k @ Wq
    k2_kernel<<<256, 256, 0, stream>>>(kmat, Wq, K2);
    // 3) W3s = scale * K2 @ W_lin  (+ bf16 copy)
    w3_kernel<<<512, 256, 0, stream>>>(K2, Wlin, W3s, W3bf);
    // 4) c[v]
    cvec_kernel<<<1, 128, 0, stream>>>(K2, kmat, b_lin, bq, cvec);
    // 5) S_enc = enc @ W3s[:, :512]^T     (WMMA + TDM, 8192x128x512)
    gemm_bf16_128<<<BB / 128, 256, 0, stream>>>(enc, W3bf, KDIM, 0,  Senc);
    // 6) S_tok = emb @ W3s[:, 512:]^T     (WMMA + TDM, 128x128x512)
    gemm_bf16_128<<<1,        256, 0, stream>>>(emb, W3bf, KDIM, DD, Stok);
    // 7) Fused mask-stream (async double-buffered) + logsumexp + gather + reduce
    finalize_kernel<<<BB, 256, 0, stream>>>(Senc, Stok, cvec, mask, parent, targ, out);
}
